// GraphAttentionNet_12893491822682
// MI455X (gfx1250) — compile-verified, matrix operand order checked by reading, not run on hardware
//
#include <hip/hip_runtime.h>
#include <hip/hip_bf16.h>
#include <stdint.h>

// ---------------- problem constants (from reference) ----------------
#define NND   50000
#define NE    1600000
#define NET   (NE + NND)      // edges incl. self-loops
#define F_IN  128
#define F1    128             // HEADS*HID
#define HEADS 4
#define HID   32
#define F2    16
#define LRELU 0.2f
#define LN_EPS 1e-5f

typedef __attribute__((ext_vector_type(16))) __bf16 v16bf;
typedef __attribute__((ext_vector_type(8)))  float  v8f;

union FragAB { v16bf v; uint4 q[2]; };

// round-to-nearest bf16 split: f ~= hi + lo (both bf16), error ~2^-17 rel
__device__ __forceinline__ void bf16_split(float f, unsigned short& hi, unsigned short& lo) {
    unsigned u  = __float_as_uint(f);
    unsigned rh = u + 0x7FFFu + ((u >> 16) & 1u);
    hi = (unsigned short)(rh >> 16);
    float fhi = __uint_as_float(((unsigned)hi) << 16);
    float r   = f - fhi;
    unsigned u2 = __float_as_uint(r);
    unsigned rl = u2 + 0x7FFFu + ((u2 >> 16) & 1u);
    lo = (unsigned short)(rl >> 16);
}

// order-preserving float <-> uint mapping for atomicMax on floats
__device__ __forceinline__ unsigned fkey(float f) {
    unsigned u = __float_as_uint(f);
    return (u & 0x80000000u) ? ~u : (u | 0x80000000u);
}
__device__ __forceinline__ float funkey(unsigned k) {
    unsigned u = (k & 0x80000000u) ? (k & 0x7FFFFFFFu) : ~k;
    return __uint_as_float(u);
}

// ---------------- init: accumulators <- bias, denoms <- 0, maxes <- -inf ----
__global__ void init_kernel(float* __restrict__ out1, float* __restrict__ out2,
                            const float* __restrict__ bias1, const float* __restrict__ bias2,
                            float* __restrict__ denom1, unsigned* __restrict__ emax1,
                            float* __restrict__ denom2, unsigned* __restrict__ emax2) {
    size_t i = (size_t)blockIdx.x * blockDim.x + threadIdx.x;
    const size_t total = (size_t)NND * F1;
    if (i >= total) return;
    out1[i] = bias1[i & 127];
    if (i < (size_t)NND * F2)    out2[i] = bias2[i & 15];
    if (i < (size_t)NND * HEADS) { denom1[i] = 0.f; emax1[i] = fkey(-3.4e38f); }
    if (i < (size_t)NND)         { denom2[i] = 0.f; emax2[i] = fkey(-3.4e38f); }
}

// ---------------- fp32 -> split bf16 (element-wise) ----------------
__global__ void split_f32_kernel(const float* __restrict__ in,
                                 unsigned short* __restrict__ hi,
                                 unsigned short* __restrict__ lo, long n) {
    long i = (long)blockIdx.x * blockDim.x + threadIdx.x;
    if (i >= n) return;
    unsigned short h, l;
    bf16_split(in[i], h, l);
    hi[i] = h; lo[i] = l;
}

// W[K,Nn] (row-major, row=K) -> T[Nn,K] split-bf16
__global__ void transpose_split_kernel(const float* __restrict__ W,
                                       unsigned short* __restrict__ thi,
                                       unsigned short* __restrict__ tlo,
                                       int K, int Nn) {
    int t = blockIdx.x * blockDim.x + threadIdx.x;
    if (t >= K * Nn) return;
    int k = t / Nn, n = t % Nn;
    unsigned short h, l;
    bf16_split(W[(size_t)k * Nn + n], h, l);
    thi[(size_t)n * K + k] = h;
    tlo[(size_t)n * K + k] = l;
}

// ---------------- split-bf16 WMMA GEMM: C[M,Nn] = A[M,K] * Bt[Nn,K]^T --------
// M%16==0, Nn%16==0, K%32==0.  One wave per 16x16 output tile.
__global__ void wmma_gemm_split(const unsigned short* __restrict__ Ahi,
                                const unsigned short* __restrict__ Alo,
                                const unsigned short* __restrict__ Bhi,
                                const unsigned short* __restrict__ Blo,
                                float* __restrict__ C, int M, int Nn, int K) {
    const int wave   = threadIdx.x >> 5;
    const int nWaves = blockDim.x >> 5;
    const int lane   = threadIdx.x & 31;
    const int l      = lane & 15;
    const int hf     = lane >> 4;          // which half-wave
    const int mBase  = blockIdx.x << 4;
    if (mBase >= M) return;
    const unsigned short* aRowH = Ahi + (size_t)(mBase + l) * K;
    const unsigned short* aRowL = Alo + (size_t)(mBase + l) * K;
    for (int nTile = wave; (nTile << 4) < Nn; nTile += nWaves) {
        const int nBase = nTile << 4;
        const unsigned short* bRowH = Bhi + (size_t)(nBase + l) * K;
        const unsigned short* bRowL = Blo + (size_t)(nBase + l) * K;
        v8f acc = {};
        for (int k = 0; k < K; k += 32) {
            const int k1 = k + (hf << 3);       // K 0..7 / 8..15
            const int k2 = k + 16 + (hf << 3);  // K 16..23 / 24..31
            FragAB ah, al, bh, bl;
            ah.q[0] = *(const uint4*)(aRowH + k1); ah.q[1] = *(const uint4*)(aRowH + k2);
            al.q[0] = *(const uint4*)(aRowL + k1); al.q[1] = *(const uint4*)(aRowL + k2);
            bh.q[0] = *(const uint4*)(bRowH + k1); bh.q[1] = *(const uint4*)(bRowH + k2);
            bl.q[0] = *(const uint4*)(bRowL + k1); bl.q[1] = *(const uint4*)(bRowL + k2);
            acc = __builtin_amdgcn_wmma_f32_16x16x32_bf16(false, ah.v, false, bh.v, (short)0, acc, false, false);
            acc = __builtin_amdgcn_wmma_f32_16x16x32_bf16(false, ah.v, false, bl.v, (short)0, acc, false, false);
            acc = __builtin_amdgcn_wmma_f32_16x16x32_bf16(false, al.v, false, bh.v, (short)0, acc, false, false);
        }
        // C/D layout: VGPR r -> (M = mBase + r + 8*hf, N = nBase + l)
        float* cp = C + (size_t)(mBase + (hf << 3)) * Nn + nBase + l;
        #pragma unroll
        for (int r = 0; r < 8; ++r) cp[(size_t)r * Nn] = acc[r];
    }
}

// ---------------- layer-1 attention scores: a = einsum(nhc,hc->nh) ----------
__global__ void attn_scores_l1(const float* __restrict__ h, const float* __restrict__ asw,
                               const float* __restrict__ adw,
                               float* __restrict__ as_out, float* __restrict__ ad_out) {
    const int n = blockIdx.x;
    const int t = threadIdx.x;                 // 0..127 ; wave == head (wave32)
    float v  = h[(size_t)n * F1 + t];
    float ps = v * asw[t];
    float pd = v * adw[t];
    #pragma unroll
    for (int off = 16; off > 0; off >>= 1) {
        ps += __shfl_xor(ps, off);
        pd += __shfl_xor(pd, off);
    }
    if ((t & 31) == 0) {
        int hd = t >> 5;
        as_out[(size_t)n * HEADS + hd] = ps;
        ad_out[(size_t)n * HEADS + hd] = pd;
    }
}

__device__ __forceinline__ void edge_sd(const int* __restrict__ ei, int e, int& s, int& d) {
    if (e < NE) { s = ei[e]; d = ei[NE + e]; } else { s = e - NE; d = s; }
}

// ---------------- layer-1 edge passes ----------------
__global__ void edge1_max(const int* __restrict__ ei, const float* __restrict__ as1,
                          const float* __restrict__ ad1, float4* __restrict__ ebuf,
                          unsigned* __restrict__ emax) {
    int e = blockIdx.x * blockDim.x + threadIdx.x;
    if (e >= NET) return;
    int s, d; edge_sd(ei, e, s, d);
    float4 a = *(const float4*)(as1 + (size_t)s * 4);
    float4 b = *(const float4*)(ad1 + (size_t)d * 4);
    float4 v;
    v.x = a.x + b.x; v.y = a.y + b.y; v.z = a.z + b.z; v.w = a.w + b.w;
    v.x = v.x > 0.f ? v.x : LRELU * v.x;
    v.y = v.y > 0.f ? v.y : LRELU * v.y;
    v.z = v.z > 0.f ? v.z : LRELU * v.z;
    v.w = v.w > 0.f ? v.w : LRELU * v.w;
    ebuf[e] = v;
    atomicMax(&emax[(size_t)d * 4 + 0], fkey(v.x));
    atomicMax(&emax[(size_t)d * 4 + 1], fkey(v.y));
    atomicMax(&emax[(size_t)d * 4 + 2], fkey(v.z));
    atomicMax(&emax[(size_t)d * 4 + 3], fkey(v.w));
}

__global__ void edge1_sum(const int* __restrict__ ei, float4* __restrict__ ebuf,
                          const unsigned* __restrict__ emax, float* __restrict__ denom) {
    int e = blockIdx.x * blockDim.x + threadIdx.x;
    if (e >= NET) return;
    int s, d; edge_sd(ei, e, s, d); (void)s;
    float4 v = ebuf[e];
    v.x = __expf(v.x - funkey(emax[(size_t)d * 4 + 0]));
    v.y = __expf(v.y - funkey(emax[(size_t)d * 4 + 1]));
    v.z = __expf(v.z - funkey(emax[(size_t)d * 4 + 2]));
    v.w = __expf(v.w - funkey(emax[(size_t)d * 4 + 3]));
    ebuf[e] = v;
    atomicAdd(&denom[(size_t)d * 4 + 0], v.x);
    atomicAdd(&denom[(size_t)d * 4 + 1], v.y);
    atomicAdd(&denom[(size_t)d * 4 + 2], v.z);
    atomicAdd(&denom[(size_t)d * 4 + 3], v.w);
}

// one wave per edge; lane j handles cols 4j..4j+3 (head = j>>3)
__global__ void edge1_aggr(const int* __restrict__ ei, const float* __restrict__ ebuf,
                           const float* __restrict__ denom, const float* __restrict__ h1,
                           float* __restrict__ out1) {
    const int lane = threadIdx.x & 31;
    const int e = (int)(((size_t)blockIdx.x * blockDim.x + threadIdx.x) >> 5);
    if (e >= NET) return;
    int s, d; edge_sd(ei, e, s, d);
    const int hd = lane >> 3;
    const float alpha = ebuf[(size_t)e * 4 + hd] / (denom[(size_t)d * 4 + hd] + 1e-16f);
    const float4 hv = *(const float4*)(h1 + (size_t)s * F1 + lane * 4);
    float* op = out1 + (size_t)d * F1 + lane * 4;
    atomicAdd(op + 0, hv.x * alpha);
    atomicAdd(op + 1, hv.y * alpha);
    atomicAdd(op + 2, hv.z * alpha);
    atomicAdd(op + 3, hv.w * alpha);
}

// ---------------- LayerNorm + ELU (in place) + split-bf16 output ------------
__global__ void ln_elu_kernel(float* __restrict__ h, const float* __restrict__ g,
                              const float* __restrict__ b,
                              unsigned short* __restrict__ hhi,
                              unsigned short* __restrict__ hlo) {
    const int lane = threadIdx.x & 31;
    const int n = (int)(((size_t)blockIdx.x * blockDim.x + threadIdx.x) >> 5);
    if (n >= NND) return;
    const size_t base = (size_t)n * F1 + lane * 4;
    float4 v = *(float4*)(h + base);
    float s = v.x + v.y + v.z + v.w;
    #pragma unroll
    for (int off = 16; off > 0; off >>= 1) s += __shfl_xor(s, off);
    const float mu = s * (1.0f / F1);
    float dx = v.x - mu, dy = v.y - mu, dz = v.z - mu, dw = v.w - mu;
    float s2 = dx * dx + dy * dy + dz * dz + dw * dw;
    #pragma unroll
    for (int off = 16; off > 0; off >>= 1) s2 += __shfl_xor(s2, off);
    const float rs = rsqrtf(s2 * (1.0f / F1) + LN_EPS);
    const float4 gg = *(const float4*)(g + lane * 4);
    const float4 bb = *(const float4*)(b + lane * 4);
    float y0 = dx * rs * gg.x + bb.x;
    float y1 = dy * rs * gg.y + bb.y;
    float y2 = dz * rs * gg.z + bb.z;
    float y3 = dw * rs * gg.w + bb.w;
    y0 = y0 > 0.f ? y0 : expm1f(y0);
    y1 = y1 > 0.f ? y1 : expm1f(y1);
    y2 = y2 > 0.f ? y2 : expm1f(y2);
    y3 = y3 > 0.f ? y3 : expm1f(y3);
    float4 y; y.x = y0; y.y = y1; y.z = y2; y.w = y3;
    *(float4*)(h + base) = y;
    unsigned short hi0, lo0, hi1, lo1, hi2, lo2, hi3, lo3;
    bf16_split(y0, hi0, lo0); bf16_split(y1, hi1, lo1);
    bf16_split(y2, hi2, lo2); bf16_split(y3, hi3, lo3);
    ushort4 uh; uh.x = hi0; uh.y = hi1; uh.z = hi2; uh.w = hi3;
    ushort4 ul; ul.x = lo0; ul.y = lo1; ul.z = lo2; ul.w = lo3;
    *(ushort4*)(hhi + base) = uh;
    *(ushort4*)(hlo + base) = ul;
}

// ---------------- layer-2 attention scores (H=1, C=16) ----------------
__global__ void attn_scores_l2(const float* __restrict__ h2, const float* __restrict__ asw,
                               const float* __restrict__ adw,
                               float* __restrict__ as2, float* __restrict__ ad2) {
    int n = blockIdx.x * blockDim.x + threadIdx.x;
    if (n >= NND) return;
    float s = 0.f, d = 0.f;
    #pragma unroll
    for (int c = 0; c < F2; ++c) {
        float v = h2[(size_t)n * F2 + c];
        s += v * asw[c];
        d += v * adw[c];
    }
    as2[n] = s; ad2[n] = d;
}

__global__ void edge2_max(const int* __restrict__ ei, const float* __restrict__ as2,
                          const float* __restrict__ ad2, float* __restrict__ ebuf,
                          unsigned* __restrict__ emax) {
    int e = blockIdx.x * blockDim.x + threadIdx.x;
    if (e >= NET) return;
    int s, d; edge_sd(ei, e, s, d);
    float v = as2[s] + ad2[d];
    v = v > 0.f ? v : LRELU * v;
    ebuf[e] = v;
    atomicMax(&emax[d], fkey(v));
}

__global__ void edge2_sum(const int* __restrict__ ei, float* __restrict__ ebuf,
                          const unsigned* __restrict__ emax, float* __restrict__ denom) {
    int e = blockIdx.x * blockDim.x + threadIdx.x;
    if (e >= NET) return;
    int s, d; edge_sd(ei, e, s, d); (void)s;
    float ee = __expf(ebuf[e] - funkey(emax[d]));
    ebuf[e] = ee;
    atomicAdd(&denom[d], ee);
}

// eighth-wave (4 lanes * float4 = 16 cols) per edge, scatter into d_out
__global__ void edge2_aggr(const int* __restrict__ ei, const float* __restrict__ ebuf,
                           const float* __restrict__ denom, const float* __restrict__ h2,
                           float* __restrict__ out) {
    long t = (long)blockIdx.x * blockDim.x + threadIdx.x;
    int e = (int)(t >> 2);
    if (e >= NET) return;
    int q = (int)(t & 3);
    int s, d; edge_sd(ei, e, s, d);
    const float alpha = ebuf[e] / (denom[d] + 1e-16f);
    const float4 hv = *(const float4*)(h2 + (size_t)s * F2 + q * 4);
    float* op = out + (size_t)d * F2 + q * 4;
    atomicAdd(op + 0, hv.x * alpha);
    atomicAdd(op + 1, hv.y * alpha);
    atomicAdd(op + 2, hv.z * alpha);
    atomicAdd(op + 3, hv.w * alpha);
}

// ======================================================================
extern "C" void kernel_launch(void* const* d_in, const int* in_sizes, int n_in,
                              void* d_out, int out_size, void* d_ws, size_t ws_size,
                              hipStream_t stream) {
    const float* x        = (const float*)d_in[0];
    const int*   ei       = (const int*)  d_in[1];
    const float* W1       = (const float*)d_in[2];
    const float* att_src1 = (const float*)d_in[3];
    const float* att_dst1 = (const float*)d_in[4];
    const float* bias1    = (const float*)d_in[5];
    const float* ln_g     = (const float*)d_in[6];
    const float* ln_b     = (const float*)d_in[7];
    const float* W2       = (const float*)d_in[8];
    const float* att_src2 = (const float*)d_in[9];
    const float* att_dst2 = (const float*)d_in[10];
    const float* bias2    = (const float*)d_in[11];
    float* out = (float*)d_out;

    // ---- workspace carve-up (256B aligned slices) ----
    char* p = (char*)d_ws;
    auto carve = [&](size_t bytes) -> char* {
        char* r = p;
        p += (bytes + 255) & ~(size_t)255;
        return r;
    };
    unsigned short* xhi    = (unsigned short*)carve((size_t)NND * F_IN * 2);
    unsigned short* xlo    = (unsigned short*)carve((size_t)NND * F_IN * 2);
    unsigned short* w1thi  = (unsigned short*)carve((size_t)F1 * F_IN * 2);
    unsigned short* w1tlo  = (unsigned short*)carve((size_t)F1 * F_IN * 2);
    unsigned short* w2thi  = (unsigned short*)carve((size_t)F2 * F1 * 2);
    unsigned short* w2tlo  = (unsigned short*)carve((size_t)F2 * F1 * 2);
    float*          h1     = (float*)carve((size_t)NND * F1 * 4);
    float*          as1    = (float*)carve((size_t)NND * HEADS * 4);
    float*          ad1    = (float*)carve((size_t)NND * HEADS * 4);
    unsigned*       emax1  = (unsigned*)carve((size_t)NND * HEADS * 4);
    float*          denom1 = (float*)carve((size_t)NND * HEADS * 4);
    float4*         ebuf1  = (float4*)carve((size_t)NET * 4 * 4);
    float*          out1   = (float*)carve((size_t)NND * F1 * 4);   // layer1 output / hact
    unsigned short* hhi    = (unsigned short*)carve((size_t)NND * F1 * 2);
    unsigned short* hlo    = (unsigned short*)carve((size_t)NND * F1 * 2);
    float*          h2     = (float*)carve((size_t)NND * F2 * 4);
    float*          as2    = (float*)carve((size_t)NND * 4);
    float*          ad2    = (float*)carve((size_t)NND * 4);
    unsigned*       emax2  = (unsigned*)carve((size_t)NND * 4);
    float*          denom2 = (float*)carve((size_t)NND * 4);
    float*          ebuf2  = (float*)carve((size_t)NET * 4);
    (void)ws_size; (void)in_sizes; (void)n_in; (void)out_size;

    const int T = 256;
    // 1) init accumulators / softmax state
    init_kernel<<<(NND * F1 + T - 1) / T, T, 0, stream>>>(out1, out, bias1, bias2,
                                                          denom1, emax1, denom2, emax2);
    // 2) convert x and weights to split-bf16 (weights pre-transposed to [N][K])
    split_f32_kernel<<<((long)NND * F_IN + T - 1) / T, T, 0, stream>>>(x, xhi, xlo, (long)NND * F_IN);
    transpose_split_kernel<<<(F_IN * F1 + T - 1) / T, T, 0, stream>>>(W1, w1thi, w1tlo, F_IN, F1);
    transpose_split_kernel<<<(F1 * F2 + T - 1) / T, T, 0, stream>>>(W2, w2thi, w2tlo, F1, F2);
    // 3) h1 = x @ W1 via split-bf16 WMMA (3125 M-tiles x 8 N-tiles, 1 wave/tile)
    wmma_gemm_split<<<NND / 16, 256, 0, stream>>>(xhi, xlo, w1thi, w1tlo, h1, NND, F1, F_IN);
    // 4) per-node attention scores (wave == head under wave32)
    attn_scores_l1<<<NND, 128, 0, stream>>>(h1, att_src1, att_dst1, as1, ad1);
    // 5-7) layer-1 edge softmax + aggregation
    edge1_max <<<(NET + T - 1) / T, T, 0, stream>>>(ei, as1, ad1, ebuf1, emax1);
    edge1_sum <<<(NET + T - 1) / T, T, 0, stream>>>(ei, ebuf1, emax1, denom1);
    edge1_aggr<<<(NET + 7) / 8, T, 0, stream>>>(ei, (const float*)ebuf1, denom1, h1, out1);
    // 8) bias (pre-folded) + LayerNorm + ELU, emit split-bf16 activations
    ln_elu_kernel<<<(NND + 7) / 8, T, 0, stream>>>(out1, ln_g, ln_b, hhi, hlo);
    // 9) h2 = act @ W2 (single 16-wide N tile -> 1 wave per block)
    wmma_gemm_split<<<NND / 16, 32, 0, stream>>>(hhi, hlo, w2thi, w2tlo, h2, NND, F2, F1);
    // 10) layer-2 scores
    attn_scores_l2<<<(NND + T - 1) / T, T, 0, stream>>>(h2, att_src2, att_dst2, as2, ad2);
    // 11-13) layer-2 edge softmax + aggregation directly into d_out (init'd to bias2)
    edge2_max <<<(NET + T - 1) / T, T, 0, stream>>>(ei, as2, ad2, ebuf2, emax2);
    edge2_sum <<<(NET + T - 1) / T, T, 0, stream>>>(ei, ebuf2, emax2, denom2);
    edge2_aggr<<<((long)NET * 4 + T - 1) / T, T, 0, stream>>>(ei, ebuf2, denom2, h2, out);
}
